// TransformerCell_73950746902727
// MI455X (gfx1250) — matrix operand order repaired and checked by reading, not compiled, not run on hardware
//
#include <hip/hip_runtime.h>

typedef __attribute__((ext_vector_type(16))) __bf16 v16bf;
typedef __attribute__((ext_vector_type(8)))  __bf16 v8bf;
typedef __attribute__((ext_vector_type(8)))  float  v8f;

#define D 512
#define LTOK 35
#define BROWS 16
#define XSTR 520   // bf16 row stride for A staging (1040B, 16B aligned, conflict-free)
#define QSTR 516   // f32 row stride
#define KVSTR 68

// ---------------- weight f32 -> bf16 conversion ----------------
__global__ void conv_bf16_kernel(const float* __restrict__ src,
                                 __bf16* __restrict__ dst, int n) {
  int i = blockIdx.x * blockDim.x + threadIdx.x;
  int stride = gridDim.x * blockDim.x;
  for (; i < n; i += stride) dst[i] = (__bf16)src[i];
}

// ---------------- 16x64 GEMM slice: acc += A(16x512,LDS bf16) * W^T slice ----
__device__ __forceinline__ void gemm_16x64(const __bf16* __restrict__ xa,
                                           const __bf16* __restrict__ W,
                                           int colBase, int m16, int sel,
                                           v8f acc[4]) {
  const __bf16* arow = xa + m16 * XSTR + sel * 16;
#pragma unroll 4
  for (int kk = 0; kk < D; kk += 32) {
    v8bf lo = *(const v8bf*)(arow + kk);
    v8bf hi = *(const v8bf*)(arow + kk + 8);
    v16bf a;
#pragma unroll
    for (int i = 0; i < 8; ++i) { a[i] = lo[i]; a[8 + i] = hi[i]; }
#pragma unroll
    for (int t2 = 0; t2 < 4; ++t2) {
      const __bf16* bp = W + (size_t)(colBase + t2 * 16 + m16) * D + kk + sel * 16;
      v16bf b = *(const v16bf*)bp;
      acc[t2] = __builtin_amdgcn_wmma_f32_16x16x32_bf16(
          false, a, false, b, (short)0, acc[t2], false, false);
    }
  }
}

// ---------------- stage one ctx token tile (16 rows x 512) into LDS bf16 -----
__device__ __forceinline__ void load_ctx_tile(
    __bf16* __restrict__ xa, int t, int b0,
    const float* __restrict__ c, const float* __restrict__ h,
    const float* __restrict__ e, const float* __restrict__ s, int tid) {
  int r = tid >> 4;
  int c0 = (tid & 15) * 32;
  int gr = b0 + r;
  const float* src;
  if (t < 32)       src = c + ((size_t)gr * 32 + t) * D;
  else if (t == 32) src = h + (size_t)gr * D;
  else if (t == 33) src = e + (size_t)gr * D;
  else              src = s + (size_t)gr * D;
  const float4* s4 = (const float4*)(src + c0);
  __bf16* dst = xa + r * XSTR + c0;
#pragma unroll
  for (int i = 0; i < 8; ++i) {
    float4 v = s4[i];
    dst[i * 4 + 0] = (__bf16)v.x;
    dst[i * 4 + 1] = (__bf16)v.y;
    dst[i * 4 + 2] = (__bf16)v.z;
    dst[i * 4 + 3] = (__bf16)v.w;
  }
}

// ---------------- fused transformer-cell kernel ----------------
__global__ void __launch_bounds__(256)
transformer_cell_kernel(const float* __restrict__ c, const float* __restrict__ h,
                        const float* __restrict__ e, const float* __restrict__ s,
                        const float* __restrict__ bq, const float* __restrict__ bk,
                        const float* __restrict__ bv, const float* __restrict__ bo,
                        const float* __restrict__ ln_a, const float* __restrict__ ln_b,
                        const __bf16* __restrict__ WqB, const __bf16* __restrict__ WkB,
                        const __bf16* __restrict__ WvB, const __bf16* __restrict__ WoB,
                        float* __restrict__ out) {
  __shared__ __bf16 Xbf[BROWS * XSTR];            // bf16 A staging
  __shared__ float  Qf[BROWS * QSTR];             // q (scaled) / later Y
  __shared__ float  KV[8 * BROWS * KVSTR];        // per-wave projection scratch
  __shared__ float  Sc[BROWS * 8 * 40];           // scores -> attn
  __shared__ float  red1[BROWS * 16], red2[BROWS * 16];
  __shared__ float  mu[BROWS], rs[BROWS];

  const int tid  = threadIdx.x;
  const int lane = tid & 31;
  const int wv   = tid >> 5;        // wave == head == 64-col slice
  const int m16  = lane & 15;
  const int sel  = lane >> 4;
  const int b0   = blockIdx.x * BROWS;
  const int colW = wv * 64;

  float bq4[4], bk4[4], bv4[4], bo4[4];
#pragma unroll
  for (int t2 = 0; t2 < 4; ++t2) {
    int col = colW + t2 * 16 + m16;
    bq4[t2] = bq[col]; bk4[t2] = bk[col]; bv4[t2] = bv[col]; bo4[t2] = bo[col];
  }

  // ---- Phase 1: Q projection (q = h @ Wq^T + bq, scaled by 1/sqrt(dk)) ----
  load_ctx_tile(Xbf, 32, b0, c, h, e, s, tid);
  __syncthreads();
  {
    v8f acc[4] = {};
    gemm_16x64(Xbf, WqB, colW, m16, sel, acc);
#pragma unroll
    for (int t2 = 0; t2 < 4; ++t2)
#pragma unroll
      for (int j = 0; j < 8; ++j)
        Qf[(j + 8 * sel) * QSTR + colW + t2 * 16 + m16] =
            (acc[t2][j] + bq4[t2]) * 0.125f;
  }
  __syncthreads();

  // ---- Phase 2: K projection per token + scores ----
  for (int t = 0; t < LTOK; ++t) {
    load_ctx_tile(Xbf, t, b0, c, h, e, s, tid);
    __syncthreads();
    v8f kacc[4] = {};
    gemm_16x64(Xbf, WkB, colW, m16, sel, kacc);
#pragma unroll
    for (int t2 = 0; t2 < 4; ++t2)
#pragma unroll
      for (int j = 0; j < 8; ++j)
        KV[(wv * BROWS + (j + 8 * sel)) * KVSTR + t2 * 16 + m16] =
            kacc[t2][j] + bk4[t2];
    __syncthreads();
    if (tid < 128) {
      int nl = tid & 15, hd = tid >> 4;
      const float* kr = &KV[(hd * BROWS + nl) * KVSTR];
      const float* qr = &Qf[nl * QSTR + hd * 64];
      float sa = 0.f;
#pragma unroll 8
      for (int d2 = 0; d2 < 64; ++d2) sa += kr[d2] * qr[d2];
      Sc[(nl * 8 + hd) * 40 + t] = sa;
    }
    __syncthreads();
  }

  // ---- Phase 3: softmax over 35 tokens per (row, head) ----
  if (tid < 128) {
    int nl = tid & 15, hd = tid >> 4;
    float* sr = &Sc[(nl * 8 + hd) * 40];
    float mx = -1e30f;
    for (int t = 0; t < LTOK; ++t) mx = fmaxf(mx, sr[t]);
    float sum = 0.f;
    for (int t = 0; t < LTOK; ++t) { float e2 = __expf(sr[t] - mx); sr[t] = e2; sum += e2; }
    float inv = 1.f / sum;
    for (int t = 0; t < LTOK; ++t) sr[t] *= inv;
  }
  __syncthreads();

  // ---- Phase 4: V projection per token, attn-weighted accumulation ----
  v8f oacc[4] = {};
  for (int t = 0; t < LTOK; ++t) {
    load_ctx_tile(Xbf, t, b0, c, h, e, s, tid);
    __syncthreads();
    v8f vacc[4] = {};
    gemm_16x64(Xbf, WvB, colW, m16, sel, vacc);
#pragma unroll
    for (int j = 0; j < 8; ++j) {
      float at = Sc[((j + 8 * sel) * 8 + wv) * 40 + t];
#pragma unroll
      for (int t2 = 0; t2 < 4; ++t2) oacc[t2][j] += at * vacc[t2][j];
    }
    __syncthreads();
  }

  // ---- Phase 5: O projection + bias + relu + layernorm ----
#pragma unroll
  for (int t2 = 0; t2 < 4; ++t2)
#pragma unroll
    for (int j = 0; j < 8; ++j)   // bv folded once (sum of attn == 1)
      Xbf[(j + 8 * sel) * XSTR + colW + t2 * 16 + m16] =
          (__bf16)(oacc[t2][j] + bv4[t2]);
  __syncthreads();
  {
    v8f yacc[4] = {};
    gemm_16x64(Xbf, WoB, colW, m16, sel, yacc);
#pragma unroll
    for (int t2 = 0; t2 < 4; ++t2)
#pragma unroll
      for (int j = 0; j < 8; ++j) {
        float y = yacc[t2][j] + bo4[t2];
        Qf[(j + 8 * sel) * QSTR + colW + t2 * 16 + m16] = y > 0.f ? y : 0.f;
      }
  }
  __syncthreads();

  // layernorm: mean/std (ddof=1), eps on std
  {
    int r = tid >> 4, jj = tid & 15;
    const float* yr = &Qf[r * QSTR + jj * 32];
    float s1 = 0.f, s2 = 0.f;
#pragma unroll 8
    for (int i = 0; i < 32; ++i) { float x = yr[i]; s1 += x; s2 += x * x; }
    red1[r * 16 + jj] = s1; red2[r * 16 + jj] = s2;
  }
  __syncthreads();
  if (tid < BROWS) {
    float a1 = 0.f, a2 = 0.f;
#pragma unroll
    for (int i = 0; i < 16; ++i) { a1 += red1[tid * 16 + i]; a2 += red2[tid * 16 + i]; }
    float mean = a1 * (1.f / 512.f);
    float var  = (a2 - a1 * a1 * (1.f / 512.f)) * (1.f / 511.f);
    var = var > 0.f ? var : 0.f;
    mu[tid] = mean;
    rs[tid] = 1.f / (sqrtf(var) + 1e-6f);
  }
  __syncthreads();
  {
    int r = tid >> 4, jj = tid & 15;
    float mean = mu[r], inv = rs[r];
    float* orow = out + (size_t)(b0 + r) * D;
#pragma unroll 8
    for (int i = 0; i < 32; ++i) {
      int col = jj * 32 + i;
      float x = Qf[r * QSTR + col];
      orow[col] = ln_a[col] * (x - mean) * inv + ln_b[col];
    }
  }
}

extern "C" void kernel_launch(void* const* d_in, const int* in_sizes, int n_in,
                              void* d_out, int out_size, void* d_ws, size_t ws_size,
                              hipStream_t stream) {
  (void)in_sizes; (void)n_in; (void)out_size; (void)ws_size;
  const float* c   = (const float*)d_in[0];
  const float* h   = (const float*)d_in[1];
  const float* e   = (const float*)d_in[2];
  const float* s   = (const float*)d_in[3];
  const float* Wq  = (const float*)d_in[4];
  const float* bq  = (const float*)d_in[5];
  const float* Wk  = (const float*)d_in[6];
  const float* bk  = (const float*)d_in[7];
  const float* Wv  = (const float*)d_in[8];
  const float* bv  = (const float*)d_in[9];
  const float* Wo  = (const float*)d_in[10];
  const float* bo  = (const float*)d_in[11];
  const float* lna = (const float*)d_in[12];
  const float* lnb = (const float*)d_in[13];
  float* out = (float*)d_out;

  const int WELEMS = D * D;                 // 262144
  __bf16* WqB = (__bf16*)d_ws;
  __bf16* WkB = WqB + WELEMS;
  __bf16* WvB = WkB + WELEMS;
  __bf16* WoB = WvB + WELEMS;

  conv_bf16_kernel<<<512, 256, 0, stream>>>(Wq, WqB, WELEMS);
  conv_bf16_kernel<<<512, 256, 0, stream>>>(Wk, WkB, WELEMS);
  conv_bf16_kernel<<<512, 256, 0, stream>>>(Wv, WvB, WELEMS);
  conv_bf16_kernel<<<512, 256, 0, stream>>>(Wo, WoB, WELEMS);

  transformer_cell_kernel<<<8192 / BROWS, 256, 0, stream>>>(
      c, h, e, s, bq, bk, bv, bo, lna, lnb, WqB, WkB, WvB, WoB, out);
}
